// MaskNet_90872918049340
// MI455X (gfx1250) — matrix operand order repaired and checked by reading, not compiled, not run on hardware
//
#include <hip/hip_runtime.h>
#include <hip/hip_bf16.h>
#include <math.h>

typedef __attribute__((ext_vector_type(16))) _Float16 v16h;
typedef __attribute__((ext_vector_type(8)))  float    v8f;

#define B_    32
#define NCTX  512
#define NAGT  64
#define D_    256
#define T_    80
#define TEMB  64
#define HDIM  32
#define MLPH  128

// ---- CDNA5 async global->LDS copy (16B per lane), ASYNCcnt-tracked --------
__device__ __forceinline__ void async_copy16(void* ldsDst, const void* gSrc) {
  unsigned l = (unsigned)(unsigned long long)
      (__attribute__((address_space(3))) char*)ldsDst;
  asm volatile("global_load_async_to_lds_b128 %0, %1, off"
               :: "v"(l), "v"(gSrc) : "memory");
}
__device__ __forceinline__ void wait_async0() {
  asm volatile("s_wait_asynccnt 0x0" ::: "memory");
}

// ---------------- elementwise f32 -> f16 ----------------
__global__ __launch_bounds__(256) void k_cvt(const float* __restrict__ src,
                                             _Float16* __restrict__ dst, int n) {
  int i = blockIdx.x * 256 + threadIdx.x;
  if (i < n) dst[i] = (_Float16)src[i];
}

// split mlp_w1 (128 x 512) into agent half / attn half (each 128 x 256), f16
__global__ __launch_bounds__(256) void k_split_w1(const float* __restrict__ w1,
                                                  _Float16* __restrict__ wa,
                                                  _Float16* __restrict__ wb) {
  int n = blockIdx.x;    // 0..127
  int k = threadIdx.x;   // 0..255
  wa[n * 256 + k] = (_Float16)w1[n * 512 + k];
  wb[n * 256 + k] = (_Float16)w1[n * 512 + 256 + k];
}

// qin = time_emb @ time_proj_w.T + b   (80 x 256, K = 64)
__global__ __launch_bounds__(256) void k_qin(const float* __restrict__ te,
                                             const float* __restrict__ w,
                                             const float* __restrict__ b,
                                             float* __restrict__ qin) {
  int t = blockIdx.x, n = threadIdx.x;
  float acc = b[n];
  #pragma unroll 8
  for (int e = 0; e < TEMB; ++e) acc += te[t * TEMB + e] * w[n * TEMB + e];
  qin[t * D_ + n] = acc;
}

// q = (qin @ wq.T + bq) * (1/sqrt(HDIM))   (80 x 256, K = 256)
__global__ __launch_bounds__(256) void k_q(const float* __restrict__ qin,
                                           const float* __restrict__ wq,
                                           const float* __restrict__ bq,
                                           float* __restrict__ q) {
  int t = blockIdx.x, n = threadIdx.x;
  float acc = bq[n];
  #pragma unroll 8
  for (int e = 0; e < D_; ++e) acc += qin[t * D_ + e] * wq[n * D_ + e];
  q[t * D_ + n] = acc * 0.17677669529663687f;
}

// ---------------- WMMA GEMM:  C[MxN] = A[MxK] @ W[NxK]^T (+bias) -----------
// grid = (N/128, M/64), block = 256 = 8 wave32 waves arranged 2(M) x 4(N).
// Each wave owns a 32x32 output (2x2 WMMA tiles, 4 accumulators) for A/B
// fragment reuse.  K stepped by 32 with double-buffered LDS filled by
// global_load_async_to_lds_b128 (ASYNCcnt) -> one s_wait_asynccnt + one
// barrier per K step, async fill of the next slab overlaps WMMA compute.
// Fragment layouts per CDNA5 ISA 7.12.2:
//   A (16x32 f16): lane m=lane&15, hi=lane>>4; half i: K = (i>>3)*16 + hi*8 + (i&7)
//   B (32x16 f16): lane n=lane&15, hi=lane>>4; half i: K = (i>>3)*16 + hi*8 + (i&7)
//   C/D (16x16 f32): lane n=lane&15; VGPR r: M = r + 8*(lane>>4)
__global__ __launch_bounds__(256) void k_gemm(const _Float16* __restrict__ A,
                                              const _Float16* __restrict__ W,
                                              const float* __restrict__ bias,
                                              float* __restrict__ C,
                                              int N, int K) {
  __shared__ __align__(16) _Float16 As[2][64 * 32];    // 2 x 4 KB
  __shared__ __align__(16) _Float16 Bs[2][128 * 32];   // 2 x 8 KB
  const int tid  = threadIdx.x;
  const int lane = tid & 31, wv = tid >> 5;
  const int wm = wv & 1, wn = wv >> 1;
  const int m = lane & 15, hi = lane >> 4;
  const int rb = blockIdx.y * 64;    // block row base (M)
  const int cb = blockIdx.x * 128;   // block col base (N)

  // cooperative staging coordinates (16B async chunks)
  const int ar = tid >> 2, ac = (tid & 3) * 8;   // A: 64 rows x 32 k
  const int br = tid >> 1, bc = (tid & 1) * 16;  // B: 128 rows x 32 k

  auto stage = [&](int buf, int k0) {
    async_copy16(&As[buf][ar * 32 + ac], A + (size_t)(rb + ar) * K + k0 + ac);
    async_copy16(&Bs[buf][br * 32 + bc], W + (size_t)(cb + br) * K + k0 + bc);
    async_copy16(&Bs[buf][br * 32 + bc + 8],
                 W + (size_t)(cb + br) * K + k0 + bc + 8);
  };

  v8f acc00 = {}, acc01 = {}, acc10 = {}, acc11 = {};
  const int nsteps = K / 32;
  stage(0, 0);
  for (int s = 0; s < nsteps; ++s) {
    const int cur = s & 1;
    wait_async0();
    __syncthreads();                       // buf[cur] ready; prev readers done
    if (s + 1 < nsteps) stage(1 - cur, (s + 1) * 32);   // overlap with compute
    v16h a0, a1, b0, b1;
    #pragma unroll
    for (int i = 0; i < 16; ++i) {
      int g = i >> 3, j = i & 7;
      int ka = g * 16 + hi * 8 + j;
      a0[i] = As[cur][(wm * 32 + m) * 32 + ka];
      a1[i] = As[cur][(wm * 32 + 16 + m) * 32 + ka];
      int kb = g * 16 + hi * 8 + j;
      b0[i] = Bs[cur][(wn * 32 + m) * 32 + kb];
      b1[i] = Bs[cur][(wn * 32 + 16 + m) * 32 + kb];
    }
    acc00 = __builtin_amdgcn_wmma_f32_16x16x32_f16(false, a0, false, b0,
                                                   (short)0, acc00, false, false);
    acc01 = __builtin_amdgcn_wmma_f32_16x16x32_f16(false, a0, false, b1,
                                                   (short)0, acc01, false, false);
    acc10 = __builtin_amdgcn_wmma_f32_16x16x32_f16(false, a1, false, b0,
                                                   (short)0, acc10, false, false);
    acc11 = __builtin_amdgcn_wmma_f32_16x16x32_f16(false, a1, false, b1,
                                                   (short)0, acc11, false, false);
  }
  const int c0 = cb + wn * 32 + m;
  const int c1 = c0 + 16;
  const float bv0 = bias ? bias[c0] : 0.0f;
  const float bv1 = bias ? bias[c1] : 0.0f;
  #pragma unroll
  for (int r = 0; r < 8; ++r) {
    int r0 = rb + wm * 32 + hi * 8 + r;
    int r1 = r0 + 16;
    C[(size_t)r0 * N + c0] = acc00[r] + bv0;
    C[(size_t)r0 * N + c1] = acc01[r] + bv1;
    C[(size_t)r1 * N + c0] = acc10[r] + bv0;
    C[(size_t)r1 * N + c1] = acc11[r] + bv1;
  }
}

// ---------------- attention (fp32 VALU; softmax needs fp32) ----------------
// grid = (B, HEADS), block = 256.  Wave handles rows t = wv, wv+8, ... (10 each)
__global__ __launch_bounds__(256) void k_attn(const float* __restrict__ q,
                                              const float* __restrict__ k,
                                              const float* __restrict__ v,
                                              float* __restrict__ ao) {
  const int b = blockIdx.x, h = blockIdx.y;
  __shared__ float qs[T_ * HDIM];
  __shared__ float sc[8][NCTX];
  const int tid = threadIdx.x, lane = tid & 31, wv = tid >> 5;
  for (int i = tid; i < T_ * HDIM; i += 256) {
    int t = i >> 5, d = i & 31;
    qs[i] = q[t * D_ + h * HDIM + d];
  }
  __syncthreads();
  const float* kb = k + ((size_t)b * NCTX) * D_ + h * HDIM;
  const float* vb = v + ((size_t)b * NCTX) * D_ + h * HDIM;
  for (int t = wv; t < T_; t += 8) {        // uniform trip count: barriers OK
    const float* qr = qs + t * HDIM;
    float vals[16];
    float mx = -3.0e38f;
    for (int ii = 0, s = lane; s < NCTX; s += 32, ++ii) {
      const float* kr = kb + (size_t)s * D_;
      float acc = 0.f;
      #pragma unroll
      for (int d = 0; d < HDIM; ++d) acc += qr[d] * kr[d];
      vals[ii] = acc;
      mx = fmaxf(mx, acc);
    }
    #pragma unroll
    for (int off = 16; off; off >>= 1) mx = fmaxf(mx, __shfl_xor(mx, off, 32));
    float sum = 0.f;
    #pragma unroll
    for (int ii = 0; ii < 16; ++ii) {
      float e = expf(vals[ii] - mx);
      sc[wv][lane + ii * 32] = e;
      sum += e;
    }
    #pragma unroll
    for (int off = 16; off; off >>= 1) sum += __shfl_xor(sum, off, 32);
    float inv = 1.f / sum;
    __syncthreads();
    float acc = 0.f;
    for (int s = 0; s < NCTX; ++s) acc += sc[wv][s] * vb[(size_t)s * D_ + lane];
    ao[((size_t)b * T_ + t) * D_ + h * HDIM + lane] = acc * inv;
    __syncthreads();
  }
}

// ---------------- fused tail: silu-MLP (WMMA) + sigmoid + mask_emb ---------
// grid = (B, N_AGENT), block = 256
__global__ __launch_bounds__(256) void k_fused(const float* __restrict__ agentp,
                                               const float* __restrict__ attnp,
                                               const float* __restrict__ b1,
                                               const _Float16* __restrict__ w2h,
                                               const float* __restrict__ b2,
                                               const float* __restrict__ w3,
                                               const float* __restrict__ b3,
                                               const float* __restrict__ mew,
                                               const float* __restrict__ meb,
                                               const float* __restrict__ esc,
                                               float* __restrict__ omask,
                                               float* __restrict__ oemb) {
  const int b = blockIdx.x, a = blockIdx.y;
  __shared__ float    agv[MLPH];
  __shared__ _Float16 h0s[T_ * MLPH];   // 20.5 KB
  __shared__ float    h1s[T_ * MLPH];   // 41 KB
  const int tid = threadIdx.x, lane = tid & 31, wv = tid >> 5;
  if (tid < MLPH)
    agv[tid] = agentp[((size_t)b * NAGT + a) * MLPH + tid] + b1[tid];
  __syncthreads();
  for (int i = tid; i < T_ * MLPH; i += 256) {
    int t = i >> 7, j = i & 127;
    float x = agv[j] + attnp[((size_t)b * T_ + t) * MLPH + j];
    h0s[i] = (_Float16)(x / (1.f + expf(-x)));        // silu
  }
  __syncthreads();
  const int m = lane & 15, hi = lane >> 4;
  // 40 tiles = 5 (M=80) x 8 (N=128); wave wv does nt==wv, mt=0..4
  for (int tt = wv; tt < 40; tt += 8) {
    const int mt = tt >> 3, nt = tt & 7;
    v8f acc = {};
    for (int k0 = 0; k0 < MLPH; k0 += 32) {
      v16h av, bv;
      #pragma unroll
      for (int i = 0; i < 16; ++i) {
        int g = i >> 3, j = i & 7;
        av[i] = h0s[(mt * 16 + m) * MLPH + k0 + g * 16 + hi * 8 + j];
        bv[i] = w2h[(size_t)(nt * 16 + m) * MLPH + k0 + g * 16 + hi * 8 + j];
      }
      acc = __builtin_amdgcn_wmma_f32_16x16x32_f16(false, av, false, bv,
                                                   (short)0, acc, false, false);
    }
    const int col = nt * 16 + m;
    const float b2v = b2[col];
    #pragma unroll
    for (int r = 0; r < 8; ++r) {
      float x = acc[r] + b2v;
      h1s[(mt * 16 + hi * 8 + r) * MLPH + col] = x / (1.f + expf(-x)); // silu
    }
  }
  __syncthreads();
  const float scale = esc[0];
  for (int t = wv; t < T_; t += 8) {
    float p = 0.f;
    #pragma unroll
    for (int jj = 0; jj < 4; ++jj)
      p += h1s[t * MLPH + lane + jj * 32] * w3[lane + jj * 32];
    #pragma unroll
    for (int off = 16; off; off >>= 1) p += __shfl_xor(p, off, 32);
    if (lane == 0) {
      float logit = p + b3[0];
      float msk = 1.f / (1.f + expf(-logit));
      size_t ridx = ((size_t)b * NAGT + a) * T_ + t;
      omask[ridx] = msk;
      #pragma unroll
      for (int c = 0; c < 4; ++c)
        oemb[ridx * 4 + c] = (msk * mew[c] + meb[c]) * scale;
    }
  }
}

// ------------------------------- host ---------------------------------------
extern "C" void kernel_launch(void* const* d_in, const int* in_sizes, int n_in,
                              void* d_out, int out_size, void* d_ws, size_t ws_size,
                              hipStream_t stream) {
  (void)in_sizes; (void)n_in; (void)out_size; (void)ws_size;
  const float* ctx    = (const float*)d_in[0];
  const float* agent  = (const float*)d_in[1];
  const float* temb   = (const float*)d_in[2];
  const float* tpw    = (const float*)d_in[3];
  const float* tpb    = (const float*)d_in[4];
  const float* ipw    = (const float*)d_in[5];
  const float* ipb    = (const float*)d_in[6];
  const float* opw    = (const float*)d_in[7];
  const float* opb    = (const float*)d_in[8];
  const float* w1     = (const float*)d_in[9];
  const float* b1     = (const float*)d_in[10];
  const float* w2     = (const float*)d_in[11];
  const float* b2     = (const float*)d_in[12];
  const float* w3     = (const float*)d_in[13];
  const float* b3     = (const float*)d_in[14];
  const float* mew    = (const float*)d_in[15];
  const float* meb    = (const float*)d_in[16];
  const float* escale = (const float*)d_in[17];

  char* ws = (char*)d_ws;
  size_t off = 0;
  auto alloc = [&](size_t bytes) -> void* {
    void* p = ws + off;
    off += (bytes + 255) & ~(size_t)255;
    return p;
  };
  float* qin     = (float*)alloc((size_t)T_ * D_ * 4);
  float* qbuf    = (float*)alloc((size_t)T_ * D_ * 4);
  float* kbuf    = (float*)alloc((size_t)B_ * NCTX * D_ * 4);
  float* vbuf    = (float*)alloc((size_t)B_ * NCTX * D_ * 4);
  float* aoraw   = (float*)alloc((size_t)B_ * T_ * D_ * 4);
  float* aobuf   = (float*)alloc((size_t)B_ * T_ * D_ * 4);
  float* attnp   = (float*)alloc((size_t)B_ * T_ * MLPH * 4);
  float* agentp  = (float*)alloc((size_t)B_ * NAGT * MLPH * 4);
  _Float16* ctx_h   = (_Float16*)alloc((size_t)B_ * NCTX * D_ * 2);
  _Float16* agent_h = (_Float16*)alloc((size_t)B_ * NAGT * D_ * 2);
  _Float16* aoraw_h = (_Float16*)alloc((size_t)B_ * T_ * D_ * 2);
  _Float16* ao_h    = (_Float16*)alloc((size_t)B_ * T_ * D_ * 2);
  _Float16* wk_h    = (_Float16*)alloc((size_t)D_ * D_ * 2);
  _Float16* wv_h    = (_Float16*)alloc((size_t)D_ * D_ * 2);
  _Float16* wop_h   = (_Float16*)alloc((size_t)D_ * D_ * 2);
  _Float16* w1a_h   = (_Float16*)alloc((size_t)MLPH * D_ * 2);
  _Float16* w1t_h   = (_Float16*)alloc((size_t)MLPH * D_ * 2);
  _Float16* w2_h    = (_Float16*)alloc((size_t)MLPH * MLPH * 2);

  auto cvt = [&](const float* s, _Float16* d, int n) {
    k_cvt<<<(n + 255) / 256, 256, 0, stream>>>(s, d, n);
  };
  // stage f16 operands
  cvt(ctx,             ctx_h,   B_ * NCTX * D_);
  cvt(agent,           agent_h, B_ * NAGT * D_);
  cvt(ipw + 1 * D_*D_, wk_h,    D_ * D_);        // wk = in_proj rows 256..511
  cvt(ipw + 2 * D_*D_, wv_h,    D_ * D_);        // wv = rows 512..767
  cvt(opw,             wop_h,   D_ * D_);
  cvt(w2,              w2_h,    MLPH * MLPH);
  k_split_w1<<<MLPH, 256, 0, stream>>>(w1, w1a_h, w1t_h);

  // q path (tiny, fp32 VALU)
  k_qin<<<T_, 256, 0, stream>>>(temb, tpw, tpb, qin);
  k_q  <<<T_, 256, 0, stream>>>(qin, ipw /*wq rows 0..255*/, ipb, qbuf);

  // K / V projections: (16384 x 256) @ (256 x 256)^T  -> WMMA
  k_gemm<<<dim3(D_ / 128, (B_ * NCTX) / 64), 256, 0, stream>>>(
      ctx_h, wk_h, ipb + 1 * D_, kbuf, D_, D_);
  k_gemm<<<dim3(D_ / 128, (B_ * NCTX) / 64), 256, 0, stream>>>(
      ctx_h, wv_h, ipb + 2 * D_, vbuf, D_, D_);

  // attention
  k_attn<<<dim3(B_, 8), 256, 0, stream>>>(qbuf, kbuf, vbuf, aoraw);

  // out_proj: (2560 x 256) @ (256 x 256)^T
  cvt(aoraw, aoraw_h, B_ * T_ * D_);
  k_gemm<<<dim3(D_ / 128, (B_ * T_) / 64), 256, 0, stream>>>(
      aoraw_h, wop_h, opb, aobuf, D_, D_);

  // attn_part: (2560 x 256) @ (128 x 256)^T ; agent_part: (2048 x 256) @ (128 x 256)^T
  cvt(aobuf, ao_h, B_ * T_ * D_);
  k_gemm<<<dim3(MLPH / 128, (B_ * T_) / 64), 256, 0, stream>>>(
      ao_h, w1t_h, nullptr, attnp, MLPH, D_);
  k_gemm<<<dim3(MLPH / 128, (B_ * NAGT) / 64), 256, 0, stream>>>(
      agent_h, w1a_h, nullptr, agentp, MLPH, D_);

  // fused silu-MLP + sigmoid + mask_emb
  float* omask = (float*)d_out;
  float* oemb  = omask + (size_t)B_ * NAGT * T_;
  k_fused<<<dim3(B_, NAGT), 256, 0, stream>>>(
      agentp, attnp, b1, w2_h, b2, w3, b3, mew, meb, escale, omask, oemb);
}
